// BiMambaEncoderLayer_V1_53541062312188
// MI455X (gfx1250) — matrix operand order repaired and checked by reading, not compiled, not run on hardware
//
#include <hip/hip_runtime.h>
#include <math.h>

// ---------------------------------------------------------------------------
// BiMamba encoder layer for gfx1250 (MI455X).
// All GEMMs run through v_wmma_f32_16x16x32_bf16 (bf16 inputs, f32 accum).
// Constants from the reference:
#define BATCH   8
#define LSEQ    1024
#define DM      512
#define ED      1024
#define DSTATE  16
#define DTR     32
#define DFF     1024
#define MROWS   (BATCH*LSEQ)     // 8192
// ---------------------------------------------------------------------------

typedef __bf16 bf16;
typedef __attribute__((ext_vector_type(16))) __bf16 bf16x16;
typedef __attribute__((ext_vector_type(8)))  __bf16 bf16x8;
typedef __attribute__((ext_vector_type(8)))  float  f32x8;

// ------------------------- fragment loaders --------------------------------
// A-matrix 16x32 bf16: lane row L%16 holds K {h*16..h*16+15} contiguously
// in elems 0..15, where h = lane>>4 (matches B fragment order).
__device__ __forceinline__ bf16x16 load_a_frag(const bf16* __restrict__ A, int ldk,
                                               int row, int kk, int half) {
  const bf16* p = A + (size_t)row * ldk + kk + half * 16;
  bf16x8 lo = *reinterpret_cast<const bf16x8*>(p);
  bf16x8 hi = *reinterpret_cast<const bf16x8*>(p + 8);
  bf16x16 r;
#pragma unroll
  for (int i = 0; i < 8; ++i) { r[i] = lo[i]; r[i + 8] = hi[i]; }
  return r;
}
// B-matrix 32x16 bf16: lane L<16 = col L holds K 0..15 contiguously,
// lanes 16..31 hold K 16..31.  W is (N,K) row-major so col n == row n of W.
__device__ __forceinline__ bf16x16 load_b_frag(const bf16* __restrict__ W, int ldk,
                                               int col, int kk, int half) {
  return *reinterpret_cast<const bf16x16*>(W + (size_t)col * ldk + kk + half * 16);
}

// ------------------------- WMMA GEMM  C = A * W^T --------------------------
// A: (M,K) bf16 row-major, W: (N,K) bf16 row-major.  One wave computes a
// 32x32 output tile (2x2 of 16x16 WMMA), looping K in steps of 32.
// EPI: 0 = f32 store, 1 = bf16 store, 2 = f32 store + bf16 copy of cols<32,
//      3 = +bias, softplus, f32   (delta), 4 = +bias, relu, bf16 (ff1),
//      5 = +bias, f32 (ff2)
template <int EPI>
__global__ __launch_bounds__(256) void gemm_wmma(
    const bf16* __restrict__ A, const bf16* __restrict__ W,
    int M, int N, int K,
    const float* __restrict__ bias,
    float* __restrict__ outF, bf16* __restrict__ outB,
    bf16* __restrict__ aux) {
  const int lane = threadIdx.x & 31;
  const int wave = threadIdx.x >> 5;
  const int half = lane >> 4;
  const int lr = lane & 15;
  const int ntile = N >> 5;
  const int tile = blockIdx.x * 8 + wave;
  if (tile >= (M >> 5) * ntile) return;
  const int m0 = (tile / ntile) * 32;
  const int n0 = (tile % ntile) * 32;

  f32x8 acc[2][2] = {};
  for (int kk = 0; kk < K; kk += 32) {
    if (kk + 32 < K) {  // gfx1250 global_prefetch_b8 for the next K panel
      __builtin_prefetch(A + (size_t)(m0 + lr) * K + kk + 32, 0, 1);
      __builtin_prefetch(W + (size_t)(n0 + lr) * K + kk + 32, 0, 1);
    }
    bf16x16 a0 = load_a_frag(A, K, m0 + lr,      kk, half);
    bf16x16 a1 = load_a_frag(A, K, m0 + 16 + lr, kk, half);
    bf16x16 b0 = load_b_frag(W, K, n0 + lr,      kk, half);
    bf16x16 b1 = load_b_frag(W, K, n0 + 16 + lr, kk, half);
    acc[0][0] = __builtin_amdgcn_wmma_f32_16x16x32_bf16(false, a0, false, b0, (short)0, acc[0][0], false, false);
    acc[0][1] = __builtin_amdgcn_wmma_f32_16x16x32_bf16(false, a0, false, b1, (short)0, acc[0][1], false, false);
    acc[1][0] = __builtin_amdgcn_wmma_f32_16x16x32_bf16(false, a1, false, b0, (short)0, acc[1][0], false, false);
    acc[1][1] = __builtin_amdgcn_wmma_f32_16x16x32_bf16(false, a1, false, b1, (short)0, acc[1][1], false, false);
  }

  // D layout: lane half h holds rows h*8+v (v=0..7), col = lane&15.
#pragma unroll
  for (int mi = 0; mi < 2; ++mi) {
#pragma unroll
    for (int ni = 0; ni < 2; ++ni) {
      const int col = n0 + ni * 16 + lr;
      const int rbase = m0 + mi * 16 + half * 8;
      float bv = 0.f;
      if constexpr (EPI == 3 || EPI == 4 || EPI == 5) bv = bias[col];
#pragma unroll
      for (int v = 0; v < 8; ++v) {
        const float val = acc[mi][ni][v];
        const size_t o = (size_t)(rbase + v) * N + col;
        if constexpr (EPI == 0) {
          outF[o] = val;
        } else if constexpr (EPI == 1) {
          outB[o] = (bf16)val;
        } else if constexpr (EPI == 2) {
          outF[o] = val;
          if (col < DTR) aux[(size_t)(rbase + v) * DTR + col] = (bf16)val;
        } else if constexpr (EPI == 3) {
          const float x = val + bv;
          outF[o] = (x > 20.f) ? x : log1pf(expf(x));   // softplus
        } else if constexpr (EPI == 4) {
          const float x = val + bv;
          outB[o] = (bf16)(x > 0.f ? x : 0.f);          // relu -> bf16
        } else if constexpr (EPI == 5) {
          outF[o] = val + bv;
        }
      }
    }
  }
}

// ------------------------- elementwise helpers -----------------------------
__global__ void cvt_bf16_kernel(const float* __restrict__ s, bf16* __restrict__ d, int n) {
  int i = blockIdx.x * 256 + threadIdx.x;
  if (i < n) d[i] = (bf16)s[i];
}

// x (B,L,DM) f32 -> bf16, optionally flipped along L (backward branch).
__global__ void cvt_x_kernel(const float* __restrict__ x, bf16* __restrict__ d, int flip) {
  int i = blockIdx.x * 256 + threadIdx.x;
  if (i >= MROWS * DM) return;
  int row = i >> 9;            // /DM
  int c = i & (DM - 1);
  int b = row >> 10;           // /LSEQ
  int l = row & (LSEQ - 1);
  int srow = flip ? (b * LSEQ + (LSEQ - 1 - l)) : row;
  d[i] = (bf16)x[(size_t)srow * DM + c];
}

// causal depthwise conv (D_CONV=4) + bias + SiLU over xz[:, 0:ED].
__global__ void conv_silu_kernel(const bf16* __restrict__ xz, const float* __restrict__ w,
                                 const float* __restrict__ cb,
                                 float* __restrict__ xcf, bf16* __restrict__ xcb) {
  const int row = blockIdx.x;            // b*L + l
  const int l = row & (LSEQ - 1);
  const int e = blockIdx.y * 256 + threadIdx.x;
  float s = cb[e];
  const float w0 = w[e * 4 + 0], w1 = w[e * 4 + 1], w2 = w[e * 4 + 2], w3 = w[e * 4 + 3];
  if (l >= 3) s += w0 * (float)xz[(size_t)(row - 3) * (2 * ED) + e];
  if (l >= 2) s += w1 * (float)xz[(size_t)(row - 2) * (2 * ED) + e];
  if (l >= 1) s += w2 * (float)xz[(size_t)(row - 1) * (2 * ED) + e];
  s += w3 * (float)xz[(size_t)row * (2 * ED) + e];
  s = s / (1.f + expf(-s));              // silu
  xcf[(size_t)row * ED + e] = s;
  xcb[(size_t)row * ED + e] = (bf16)s;
}

// selective scan: one lane per (b,e) channel, 16 states in registers,
// B/C (32 uniform floats per step) staged through LDS 8 steps at a time.
__global__ __launch_bounds__(256) void scan_kernel(
    const float* __restrict__ delta, const float* __restrict__ xc,
    const float* __restrict__ dbc, const float* __restrict__ Alog,
    const float* __restrict__ Dp, float* __restrict__ y) {
  const int b = blockIdx.x;
  const int e = blockIdx.y * 256 + threadIdx.x;
  const int t = threadIdx.x;
  float Arow[DSTATE];
#pragma unroll
  for (int n = 0; n < DSTATE; ++n) Arow[n] = -expf(Alog[(size_t)e * DSTATE + n]);
  const float Dv = Dp[e];
  float h[DSTATE];
#pragma unroll
  for (int n = 0; n < DSTATE; ++n) h[n] = 0.f;

  __shared__ float sbc[8][32];
  for (int l0 = 0; l0 < LSEQ; l0 += 8) {
    const int step = t >> 5, j = t & 31;
    sbc[step][j] = dbc[(size_t)(b * LSEQ + l0 + step) * 64 + DTR + j];
    __syncthreads();
#pragma unroll
    for (int s = 0; s < 8; ++s) {
      const size_t row = (size_t)b * LSEQ + l0 + s;
      const float dl = delta[row * ED + e];
      const float xv = xc[row * ED + e];
      const float dx = dl * xv;
      float acc = 0.f;
#pragma unroll
      for (int n = 0; n < DSTATE; ++n) {
        const float da = expf(dl * Arow[n]);
        h[n] = da * h[n] + dx * sbc[s][n];
        acc += h[n] * sbc[s][DSTATE + n];
      }
      y[row * ED + e] = acc + Dv * xv;
    }
    __syncthreads();
  }
}

// gate: g = y*silu(z) + xc*(1 - silu(z)),  z = xz[:, ED:2ED]
__global__ void gate_kernel(const float* __restrict__ y, const float* __restrict__ xc,
                            const bf16* __restrict__ xz, bf16* __restrict__ g) {
  size_t i = (size_t)blockIdx.x * 256 + threadIdx.x;
  if (i >= (size_t)MROWS * ED) return;
  const size_t row = i >> 10;          // /ED
  const int e = (int)(i & (ED - 1));
  const float z = (float)xz[row * (2 * ED) + ED + e];
  const float zz = z / (1.f + expf(-z));
  g[i] = (bf16)(y[i] * zz + xc[i] * (1.f - zz));
}

// out = LayerNorm(a + b) * g + beta   (row width DM=512, 128 threads/row)
__global__ void ln_add_kernel(const float* __restrict__ a, const float* __restrict__ b,
                              const float* __restrict__ g, const float* __restrict__ be,
                              float* __restrict__ out) {
  __shared__ float rs[128], rq[128];
  const int row = blockIdx.x, t = threadIdx.x;
  const float* pa = a + (size_t)row * DM;
  const float* pb = b + (size_t)row * DM;
  float v[4], s = 0.f, q = 0.f;
#pragma unroll
  for (int i = 0; i < 4; ++i) {
    const int idx = t + i * 128;
    const float x = pa[idx] + pb[idx];
    v[i] = x; s += x; q += x * x;
  }
  rs[t] = s; rq[t] = q; __syncthreads();
  for (int off = 64; off > 0; off >>= 1) {
    if (t < off) { rs[t] += rs[t + off]; rq[t] += rq[t + off]; }
    __syncthreads();
  }
  const float m = rs[0] * (1.f / DM);
  const float var = rq[0] * (1.f / DM) - m * m;
  const float inv = rsqrtf(var + 1e-5f);
#pragma unroll
  for (int i = 0; i < 4; ++i) {
    const int idx = t + i * 128;
    out[(size_t)row * DM + idx] = (v[i] - m) * inv * g[idx] + be[idx];
  }
}

// xs = xf + xb (f32 + bf16 copy for the FFN A-matrix)
__global__ void add_kernel(const float* __restrict__ a, const float* __restrict__ b,
                           float* __restrict__ o, bf16* __restrict__ ob, int n) {
  int i = blockIdx.x * 256 + threadIdx.x;
  if (i < n) { const float v = a[i] + b[i]; o[i] = v; ob[i] = (bf16)v; }
}

// ---------------------------------------------------------------------------
extern "C" void kernel_launch(void* const* d_in, const int* in_sizes, int n_in,
                              void* d_out, int out_size, void* d_ws, size_t ws_size,
                              hipStream_t stream) {
  (void)in_sizes; (void)n_in; (void)out_size; (void)ws_size;
  const float* x    = (const float*)d_in[28];
  const float* n1g  = (const float*)d_in[18]; const float* n1b = (const float*)d_in[19];
  const float* n2g  = (const float*)d_in[20]; const float* n2b = (const float*)d_in[21];
  const float* n3g  = (const float*)d_in[22]; const float* n3b = (const float*)d_in[23];
  const float* ff1w = (const float*)d_in[24]; const float* ff1b = (const float*)d_in[25];
  const float* ff2w = (const float*)d_in[26]; const float* ff2b = (const float*)d_in[27];

  char* base = (char*)d_ws;
  size_t cur = 0;
  auto alloc = [&](size_t bytes) -> void* {
    void* p = base + cur;
    cur = (cur + bytes + 255) & ~(size_t)255;
    return p;
  };

  // persistent + per-direction (reused) scratch
  bf16*  bx     = (bf16*)alloc((size_t)MROWS * DM * 2);         // x (maybe flipped) bf16
  bf16*  w_in   = (bf16*)alloc((size_t)2 * ED * DM * 2);
  bf16*  w_xp   = (bf16*)alloc((size_t)(DTR + 2 * DSTATE) * ED * 2);
  bf16*  w_dt   = (bf16*)alloc((size_t)ED * DTR * 2);
  bf16*  w_out  = (bf16*)alloc((size_t)DM * ED * 2);
  bf16*  w_ff1  = (bf16*)alloc((size_t)DFF * DM * 2);
  bf16*  w_ff2  = (bf16*)alloc((size_t)DM * DFF * 2);
  bf16*  xz     = (bf16*)alloc((size_t)MROWS * 2 * ED * 2);     // in_proj out
  float* xcf    = (float*)alloc((size_t)MROWS * ED * 4);        // conv+silu f32
  bf16*  xcb    = (bf16*)alloc((size_t)MROWS * ED * 2);         // conv+silu bf16
  float* dbc    = (float*)alloc((size_t)MROWS * 64 * 4);        // x_proj out
  bf16*  dbc32  = (bf16*)alloc((size_t)MROWS * DTR * 2);        // dbc[:, :32] bf16
  float* deltab = (float*)alloc((size_t)MROWS * ED * 4);        // softplus(dt)
  float* ybuf   = (float*)alloc((size_t)MROWS * ED * 4);        // scan out
  bf16*  gbuf   = (bf16*)alloc((size_t)MROWS * ED * 2);         // gated, bf16
  float* mout   = (float*)alloc((size_t)MROWS * DM * 4);        // out_proj out
  float* xfbuf  = (float*)alloc((size_t)MROWS * DM * 4);        // LN1 out
  float* xbbuf  = (float*)alloc((size_t)MROWS * DM * 4);        // LN2 out
  // FFN overlays (direction scratch is dead by then)
  float* xs  = deltab;              // 16MB into 32MB region
  bf16*  xsb = gbuf;
  bf16*  h1  = xcb;
  float* h2  = xcf;

  const int thr = 256;
  auto gblk = [](size_t n) { return (int)((n + 255) / 256); };
  auto gemm_blocks = [](int M, int N) { return ((M >> 5) * (N >> 5) + 7) / 8; };

  // FFN weights once
  cvt_bf16_kernel<<<gblk((size_t)DFF * DM), thr, 0, stream>>>(ff1w, w_ff1, DFF * DM);
  cvt_bf16_kernel<<<gblk((size_t)DM * DFF), thr, 0, stream>>>(ff2w, w_ff2, DM * DFF);

  for (int d = 0; d < 2; ++d) {
    const float* in_w   = (const float*)d_in[d ? 9  : 0];
    const float* conv_w = (const float*)d_in[d ? 10 : 1];
    const float* conv_b = (const float*)d_in[d ? 11 : 2];
    const float* xproj_w= (const float*)d_in[d ? 12 : 3];
    const float* dt_w   = (const float*)d_in[d ? 13 : 4];
    const float* dt_b   = (const float*)d_in[d ? 14 : 5];
    const float* Alog   = (const float*)d_in[d ? 15 : 6];
    const float* Dp     = (const float*)d_in[d ? 16 : 7];
    const float* out_w  = (const float*)d_in[d ? 17 : 8];

    cvt_x_kernel<<<gblk((size_t)MROWS * DM), thr, 0, stream>>>(x, bx, d);
    cvt_bf16_kernel<<<gblk((size_t)2 * ED * DM), thr, 0, stream>>>(in_w, w_in, 2 * ED * DM);
    cvt_bf16_kernel<<<gblk((size_t)64 * ED), thr, 0, stream>>>(xproj_w, w_xp, 64 * ED);
    cvt_bf16_kernel<<<gblk((size_t)ED * DTR), thr, 0, stream>>>(dt_w, w_dt, ED * DTR);
    cvt_bf16_kernel<<<gblk((size_t)DM * ED), thr, 0, stream>>>(out_w, w_out, DM * ED);

    // xz = x @ in_w^T                     (8192 x 2048, K=512) -> bf16
    gemm_wmma<1><<<gemm_blocks(MROWS, 2 * ED), thr, 0, stream>>>(
        bx, w_in, MROWS, 2 * ED, DM, nullptr, nullptr, xz, nullptr);
    // conv + SiLU
    conv_silu_kernel<<<dim3(MROWS, ED / 256), thr, 0, stream>>>(xz, conv_w, conv_b, xcf, xcb);
    // dbc = xc @ xproj_w^T                (8192 x 64, K=1024) -> f32 + bf16 cols<32
    gemm_wmma<2><<<gemm_blocks(MROWS, 64), thr, 0, stream>>>(
        xcb, w_xp, MROWS, 64, ED, nullptr, dbc, nullptr, dbc32);
    // delta = softplus(dbc[:, :32] @ dt_w^T + dt_b)   (8192 x 1024, K=32)
    gemm_wmma<3><<<gemm_blocks(MROWS, ED), thr, 0, stream>>>(
        dbc32, w_dt, MROWS, ED, DTR, dt_b, deltab, nullptr, nullptr);
    // selective scan
    scan_kernel<<<dim3(BATCH, ED / 256), thr, 0, stream>>>(deltab, xcf, dbc, Alog, Dp, ybuf);
    // gating
    gate_kernel<<<gblk((size_t)MROWS * ED), thr, 0, stream>>>(ybuf, xcf, xz, gbuf);
    // mout = g @ out_w^T                  (8192 x 512, K=1024) -> f32
    gemm_wmma<0><<<gemm_blocks(MROWS, DM), thr, 0, stream>>>(
        gbuf, w_out, MROWS, DM, ED, nullptr, mout, nullptr, nullptr);
    // xf/xb = LN(mout + x)
    ln_add_kernel<<<MROWS, 128, 0, stream>>>(mout, x, d ? n2g : n1g, d ? n2b : n1b,
                                             d ? xbbuf : xfbuf);
  }

  // FFN: xs = xf + xb ; h = relu(xs@ff1^T + b1)@ff2^T + b2 ; out = LN(h + xs)
  add_kernel<<<gblk((size_t)MROWS * DM), thr, 0, stream>>>(xfbuf, xbbuf, xs, xsb, MROWS * DM);
  gemm_wmma<4><<<gemm_blocks(MROWS, DFF), thr, 0, stream>>>(
      xsb, w_ff1, MROWS, DFF, DM, ff1b, nullptr, h1, nullptr);
  gemm_wmma<5><<<gemm_blocks(MROWS, DM), thr, 0, stream>>>(
      h1, w_ff2, MROWS, DM, DFF, ff2b, h2, nullptr, nullptr);
  ln_add_kernel<<<MROWS, 128, 0, stream>>>(h2, xs, n3g, n3b, (float*)d_out);
}
